// LossComputerDINO_76184129896813
// MI455X (gfx1250) — matrix operand order repaired and checked by reading, not compiled, not run on hardware
//
#include <hip/hip_runtime.h>
#include <hip/hip_bf16.h>
#include <math.h>

typedef __bf16 bf16_t;
typedef __attribute__((ext_vector_type(16))) __bf16 v16bf;
typedef __attribute__((ext_vector_type(8)))  __bf16 v8bf;
typedef __attribute__((ext_vector_type(4)))  __bf16 v4bf;
typedef __attribute__((ext_vector_type(8)))  float  v8f;

#define D_DIM   65536
#define N_T     256
#define N_G     256
#define N_L     1024
#define N_S     1280
#define BS_TILE 64
#define K_CHUNK 4096
#define NUM_KCHUNK (D_DIM / K_CHUNK)   // 16
#define NUM_STILE  (N_S / BS_TILE)     // 20
#define B_STRIDE   40                  // 32 bf16 + 8 pad (80B row, 16B aligned, conflict-free)

// ---------------------------------------------------------------------------
// Kernel 1: student per-row logsumexp  M_s = logsumexp(x/ts).
// Also zeroes the two global accumulators (block 0) before the matmul runs.
// ---------------------------------------------------------------------------
__global__ __launch_bounds__(256) void dino_student_stats(
    const float* __restrict__ sg, const float* __restrict__ sl,
    const float* __restrict__ temp_s, float* __restrict__ lse_out,
    float* __restrict__ sum_all, float* __restrict__ sum_diag)
{
  const int row = blockIdx.x;
  const int tid = threadIdx.x;
  if (row == 0 && tid == 0) { *sum_all = 0.0f; *sum_diag = 0.0f; }
  const float* x = (row < N_G) ? (sg + (size_t)row * D_DIM)
                               : (sl + (size_t)(row - N_G) * D_DIM);
  const float invT = 1.0f / temp_s[0];
  float m = -INFINITY, s = 0.0f;
  const float4* xv = (const float4*)x;
  for (int i = tid; i < D_DIM / 4; i += 256) {
    float4 v = xv[i];
    float y0 = v.x * invT, y1 = v.y * invT, y2 = v.z * invT, y3 = v.w * invT;
    float lm = fmaxf(fmaxf(y0, y1), fmaxf(y2, y3));
    float e  = __expf(y0 - lm) + __expf(y1 - lm) + __expf(y2 - lm) + __expf(y3 - lm);
    float M  = fmaxf(m, lm);
    s = s * __expf(m - M) + e * __expf(lm - M);
    m = M;
  }
  __shared__ float sm[256], ss[256];
  sm[tid] = m; ss[tid] = s;
  __syncthreads();
  for (int off = 128; off > 0; off >>= 1) {
    if (tid < off) {
      float m2 = sm[tid + off], s2 = ss[tid + off];
      float M  = fmaxf(sm[tid], m2);
      ss[tid]  = ss[tid] * __expf(sm[tid] - M) + s2 * __expf(m2 - M);
      sm[tid]  = M;
    }
    __syncthreads();
  }
  if (tid == 0) lse_out[row] = sm[0] + __logf(ss[0]);
}

// ---------------------------------------------------------------------------
// Kernel 2: teacher probs  p = softmax((x - center)/tt), materialized as bf16
// (33.5 MB -> lives in the 192MB L2 for the whole matmul phase).
// ---------------------------------------------------------------------------
__global__ __launch_bounds__(256) void dino_teacher_prob(
    const float* __restrict__ xt, const float* __restrict__ center,
    const float* __restrict__ temp_t, bf16_t* __restrict__ probT)
{
  const int row = blockIdx.x;
  const int tid = threadIdx.x;
  const float* x = xt + (size_t)row * D_DIM;
  const float invT = 1.0f / temp_t[0];
  const float4* xv = (const float4*)x;
  const float4* cv = (const float4*)center;
  float m = -INFINITY, s = 0.0f;
  for (int i = tid; i < D_DIM / 4; i += 256) {
    float4 v = xv[i]; float4 c = cv[i];
    float y0 = (v.x - c.x) * invT, y1 = (v.y - c.y) * invT;
    float y2 = (v.z - c.z) * invT, y3 = (v.w - c.w) * invT;
    float lm = fmaxf(fmaxf(y0, y1), fmaxf(y2, y3));
    float e  = __expf(y0 - lm) + __expf(y1 - lm) + __expf(y2 - lm) + __expf(y3 - lm);
    float M  = fmaxf(m, lm);
    s = s * __expf(m - M) + e * __expf(lm - M);
    m = M;
  }
  __shared__ float sm[256], ss[256];
  sm[tid] = m; ss[tid] = s;
  __syncthreads();
  for (int off = 128; off > 0; off >>= 1) {
    if (tid < off) {
      float m2 = sm[tid + off], s2 = ss[tid + off];
      float M  = fmaxf(sm[tid], m2);
      ss[tid]  = ss[tid] * __expf(sm[tid] - M) + s2 * __expf(m2 - M);
      sm[tid]  = M;
    }
    __syncthreads();
  }
  const float Mrow = sm[0];
  const float invS = 1.0f / ss[0];
  bf16_t* dst = probT + (size_t)row * D_DIM;
  for (int i = tid; i < D_DIM / 4; i += 256) {
    float4 v = xv[i]; float4 c = cv[i];
    v4bf p;
    p[0] = (__bf16)(__expf((v.x - c.x) * invT - Mrow) * invS);
    p[1] = (__bf16)(__expf((v.y - c.y) * invT - Mrow) * invS);
    p[2] = (__bf16)(__expf((v.z - c.z) * invT - Mrow) * invS);
    p[3] = (__bf16)(__expf((v.w - c.w) * invT - Mrow) * invS);
    *(v4bf*)(dst + 4 * (size_t)i) = p;
  }
}

// ---------------------------------------------------------------------------
// Kernel 3: split-K WMMA matmul.  Per WG: 256 teacher rows x 64 student rows
// over a 4096-wide K chunk. Student f32 -> bf16 staged through LDS (read from
// HBM exactly once overall); teacher bf16 read direct (L2-resident).
// Each WG reduces its accumulators to (sum_all, sum_diag) scalars.
// ---------------------------------------------------------------------------
__global__ __launch_bounds__(256) void dino_matmul(
    const float* __restrict__ sg, const float* __restrict__ sl,
    const bf16_t* __restrict__ probT,
    float* __restrict__ sum_all, float* __restrict__ sum_diag)
{
  __shared__ __attribute__((aligned(16))) bf16_t Bt[BS_TILE * B_STRIDE];
  __shared__ float redA[256], redD[256];

  const int tid  = threadIdx.x;
  const int wave = tid >> 5;
  const int lane = tid & 31;
  const int hi   = lane >> 4;     // half-lane select
  const int ln16 = lane & 15;
  const int sBase  = blockIdx.x * BS_TILE;
  const size_t kBase = (size_t)blockIdx.y * K_CHUNK;

  // 64-row student tile never straddles the global/local boundary (64 | 256).
  const float* sPtr = (sBase < N_G) ? (sg + (size_t)sBase * D_DIM)
                                    : (sl + (size_t)(sBase - N_G) * D_DIM);
  // staging: thread -> (row = tid/4, 8 consecutive k at (tid%4)*8)
  const int srow = tid >> 2;
  const int scol = (tid & 3) * 8;
  const float* sLoad  = sPtr + (size_t)srow * D_DIM + kBase + scol;
  bf16_t*      sStore = Bt + srow * B_STRIDE + scol;

  // A operand: lane ln16 = teacher row, half-lanes split K per ISA layout
  const bf16_t* aPtr0 = probT + (size_t)(wave * 32 + ln16) * D_DIM + kBase + hi * 8;
  const bf16_t* aPtr1 = aPtr0 + (size_t)16 * D_DIM;

  v8f acc[2][4] = {};

  for (int kk = 0; kk < K_CHUNK; kk += 32) {
    // ---- stage student B tile (64 x 32) f32 -> bf16 into LDS ----
    float4 f0 = *(const float4*)(sLoad + kk);
    float4 f1 = *(const float4*)(sLoad + kk + 4);
    v8bf bb;
    bb[0] = (__bf16)f0.x; bb[1] = (__bf16)f0.y; bb[2] = (__bf16)f0.z; bb[3] = (__bf16)f0.w;
    bb[4] = (__bf16)f1.x; bb[5] = (__bf16)f1.y; bb[6] = (__bf16)f1.z; bb[7] = (__bf16)f1.w;
    __syncthreads();                 // previous iteration's LDS reads done
    *(v8bf*)sStore = bb;
    __syncthreads();                 // stores visible

    // ---- load A tiles (2 x 16x32 bf16) direct from L2-resident probT ----
    v8bf a0lo = *(const v8bf*)(aPtr0 + kk);
    v8bf a0hi = *(const v8bf*)(aPtr0 + kk + 16);
    v8bf a1lo = *(const v8bf*)(aPtr1 + kk);
    v8bf a1hi = *(const v8bf*)(aPtr1 + kk + 16);
    v16bf a0 = __builtin_shufflevector(a0lo, a0hi, 0,1,2,3,4,5,6,7,8,9,10,11,12,13,14,15);
    v16bf a1 = __builtin_shufflevector(a1lo, a1hi, 0,1,2,3,4,5,6,7,8,9,10,11,12,13,14,15);

#pragma unroll
    for (int st = 0; st < 4; ++st) {
      // B lane ln16 = student column, half-lanes split K per ISA layout (same as A)
      const bf16_t* bp = Bt + (st * 16 + ln16) * B_STRIDE + hi * 8;
      v8bf blo = *(const v8bf*)bp;
      v8bf bhi = *(const v8bf*)(bp + 16);
      v16bf b = __builtin_shufflevector(blo, bhi, 0,1,2,3,4,5,6,7,8,9,10,11,12,13,14,15);
      acc[0][st] = __builtin_amdgcn_wmma_f32_16x16x32_bf16(
          false, a0, false, b, (short)0, acc[0][st], false, false);
      acc[1][st] = __builtin_amdgcn_wmma_f32_16x16x32_bf16(
          false, a1, false, b, (short)0, acc[1][st], false, false);
    }
  }

  // ---- reduce accumulators: total sum + diagonal (t == s) sum ----
  float pAll = 0.0f, pDiag = 0.0f;
#pragma unroll
  for (int tt = 0; tt < 2; ++tt)
#pragma unroll
    for (int st = 0; st < 4; ++st)
#pragma unroll
      for (int v = 0; v < 8; ++v) {
        float e = acc[tt][st][v];
        pAll += e;
        int gt = wave * 32 + tt * 16 + hi * 8 + v;   // C/D layout: M = v + 8*hi
        int gs = sBase + st * 16 + ln16;             //            N = lane%16
        if (gt == gs) pDiag += e;
      }
  redA[tid] = pAll; redD[tid] = pDiag;
  __syncthreads();
  for (int off = 128; off > 0; off >>= 1) {
    if (tid < off) { redA[tid] += redA[tid + off]; redD[tid] += redD[tid + off]; }
    __syncthreads();
  }
  if (tid == 0) { atomicAdd(sum_all, redA[0]); atomicAdd(sum_diag, redD[0]); }
}

// ---------------------------------------------------------------------------
// Kernel 4: fold the logsumexp corrections and normalize.
// ce[t,s] = M_s - dot(p_t, x_s)/ts  =>
// total = N_T*sum(M_s) - sum_{t<256} M_t - (sum_all - sum_diag)/ts
// ---------------------------------------------------------------------------
__global__ __launch_bounds__(256) void dino_finalize(
    const float* __restrict__ lse, const float* __restrict__ sums,
    const float* __restrict__ temp_s, float* __restrict__ out)
{
  __shared__ float s1[256], s2[256];
  const int tid = threadIdx.x;
  float a = 0.0f, d = 0.0f;
  for (int i = tid; i < N_S; i += 256) {
    float m = lse[i];
    a += m;
    if (i < N_T) d += m;
  }
  s1[tid] = a; s2[tid] = d;
  __syncthreads();
  for (int off = 128; off > 0; off >>= 1) {
    if (tid < off) { s1[tid] += s1[tid + off]; s2[tid] += s2[tid + off]; }
    __syncthreads();
  }
  if (tid == 0) {
    float invTs  = 1.0f / temp_s[0];
    float Mterm  = (float)N_T * s1[0] - s2[0];
    float dotterm = (sums[0] - sums[1]) * invTs;
    float nterms = (float)(N_T * N_S - N_T);
    out[0] = (Mterm - dotterm) / nterms;
  }
}

// ---------------------------------------------------------------------------
extern "C" void kernel_launch(void* const* d_in, const int* in_sizes, int n_in,
                              void* d_out, int out_size, void* d_ws, size_t ws_size,
                              hipStream_t stream) {
  (void)in_sizes; (void)n_in; (void)out_size; (void)ws_size;
  const float* sg     = (const float*)d_in[0];
  const float* sl     = (const float*)d_in[1];
  const float* xt     = (const float*)d_in[2];
  const float* center = (const float*)d_in[3];
  const float* temp_s = (const float*)d_in[4];
  const float* temp_t = (const float*)d_in[5];

  char* ws = (char*)d_ws;
  bf16_t* probT = (bf16_t*)ws;                                   // 33,554,432 B
  float*  lse   = (float*)(ws + (size_t)N_T * D_DIM * 2);        // 1280 * 4 B
  float*  sums  = (float*)(ws + (size_t)N_T * D_DIM * 2 + 8192); // 2 floats
  float*  out   = (float*)d_out;

  dino_student_stats<<<N_S, 256, 0, stream>>>(sg, sl, temp_s, lse, sums, sums + 1);
  dino_teacher_prob <<<N_T, 256, 0, stream>>>(xt, center, temp_t, probT);
  dino_matmul<<<dim3(NUM_STILE, NUM_KCHUNK), 256, 0, stream>>>(sg, sl, probT, sums, sums + 1);
  dino_finalize<<<1, 256, 0, stream>>>(lse, sums, temp_s, out);
}
